// DiTBlock_GQA_70815420776979
// MI455X (gfx1250) — compile-verified, matrix operand order checked by reading, not run on hardware
//
#include <hip/hip_runtime.h>
#include <hip/hip_bf16.h>

typedef __attribute__((ext_vector_type(16))) __bf16 bf16x16;
typedef __attribute__((ext_vector_type(8)))  float  f32x8;
typedef __attribute__((ext_vector_type(4)))  unsigned int u32x4;

#define B_    2
#define N_    2048
#define D_    1024
#define HQ_   16
#define HKV_  4
#define HD_   64
#define MTOK  (B_ * N_)        // 4096 token rows
#define ADA_LD (6 * D_)        // 6144

__device__ __forceinline__ __bf16 tobf(float f) { return (__bf16)f; }

// Load a 16-bit WMMA fragment (A-style layout, also used for our transposed-B
// operands).  Per the CDNA5 ISA 16-bit A 16x32 layout:
//   halves 0..7  -> K = k0 + (lane>=16 ? 8 : 0) + h
//   halves 8..15 -> K = 16 + same
// Caller passes p already offset by row*ld + k0 + ((lane>>4)<<3).
__device__ __forceinline__ bf16x16 load_frag(const __bf16* p) {
    union { bf16x16 v; u32x4 u[2]; } r;
    r.u[0] = *(const u32x4*)(p);
    r.u[1] = *(const u32x4*)(p + 16);
    return r.v;
}

__device__ __forceinline__ f32x8 wmma_bf16(bf16x16 a, bf16x16 b, f32x8 c) {
    return __builtin_amdgcn_wmma_f32_16x16x32_bf16(
        false, a, false, b, (short)0, c, false, false);
}

// CDNA5 async DMA: global -> LDS, tracked on ASYNCcnt (no VGPR staging).
__device__ __forceinline__ void async_load_b128(unsigned lds_off,
                                                const __bf16* gaddr) {
    asm volatile("global_load_async_to_lds_b128 %0, %1, off"
                 :: "v"(lds_off), "v"(gaddr) : "memory");
}
__device__ __forceinline__ void wait_async_le(int n) {
    if (n == 0)
        asm volatile("s_wait_asynccnt 0x0" ::: "memory");
    else
        asm volatile("s_wait_asynccnt 0x1" ::: "memory");
}

// ---------------------------------------------------------------- ada GEMV
__global__ void k_ada(const float* __restrict__ t_emb,
                      const float* __restrict__ Wada,
                      const float* __restrict__ bada,
                      float* __restrict__ ada) {
    int idx = blockIdx.x * 256 + threadIdx.x;   // B*6D = 12288 threads
    int b = idx / ADA_LD, c = idx % ADA_LD;
    float acc = bada[c];
    for (int k = 0; k < D_; ++k) {
        float t = t_emb[b * D_ + k];
        float s = t / (1.0f + __expf(-t));      // silu
        acc += s * Wada[(size_t)k * ADA_LD + c];
    }
    ada[idx] = acc;
}

// ------------------------------------------- weight fp32 -> bf16 transpose
__global__ void k_cvt_t(const float* __restrict__ W, __bf16* __restrict__ Wt,
                        int K, int Nc) {
    int idx = blockIdx.x * 256 + threadIdx.x;
    if (idx >= K * Nc) return;
    int k = idx / Nc, n = idx % Nc;
    Wt[(size_t)n * K + k] = tobf(W[idx]);
}

// --------------------------------------- RMSNorm + adaLN modulate -> bf16
__global__ void k_norm_mod(const float* __restrict__ x,
                           const float* __restrict__ g,
                           const float* __restrict__ ada,
                           int scale_off, int shift_off,
                           __bf16* __restrict__ xn) {
    __shared__ float red[256];
    int bid = blockIdx.x;            // token row in [0, MTOK)
    int b = bid / N_;
    const float* row = x + (size_t)bid * D_;
    int t = threadIdx.x;
    float ss = 0.f;
    for (int c = t; c < D_; c += 256) { float v = row[c]; ss += v * v; }
    red[t] = ss;
    __syncthreads();
    for (int s = 128; s > 0; s >>= 1) {
        if (t < s) red[t] += red[t + s];
        __syncthreads();
    }
    float rinv = rsqrtf(red[0] / (float)D_ + 1e-6f);
    const float* adab = ada + (size_t)b * ADA_LD;
    __bf16* orow = xn + (size_t)bid * D_;
    for (int c = t; c < D_; c += 256) {
        float v = row[c] * rinv * g[c] * (1.0f + adab[scale_off + c]) +
                  adab[shift_off + c];
        orow[c] = tobf(v);
    }
}

// -------------------------------------------------- generic bf16 WMMA GEMM
// C[M,Nc] = A[M,K] @ W[K,Nc], W given transposed (Wt[Nc,K]).
// B tile (64 cols x 32 k) is staged in LDS via async DMA, double-buffered.
// mode 0: out f32      mode 2: gelu(out+bias) -> bf16
// mode 3: resid + gate*(out[+bias]) -> f32
__global__ void k_gemm(const __bf16* __restrict__ A,
                       const __bf16* __restrict__ Wt,
                       int K, int Nc,
                       float* __restrict__ outF, __bf16* __restrict__ outB,
                       const float* __restrict__ bias,
                       const float* __restrict__ resid,
                       const float* __restrict__ ada, int gate_off,
                       int mode) {
    __shared__ __align__(16) __bf16 ldsB[2][64 * 32];   // 2 x 4 KB

    int tid  = threadIdx.x;
    int lane = tid & 31;
    int wave = tid >> 5;
    int mrow0 = (blockIdx.y * 8 + wave) * 16;
    int n0 = blockIdx.x * 64;
    int l16 = lane & 15;
    int hi8 = (lane >> 4) << 3;

    // async-copy assignment: thread t moves 16B: col = t>>2, k-chunk = (t&3)*8
    int bcol = tid >> 2;
    int bkq  = (tid & 3) << 3;
    const __bf16* gB = Wt + (size_t)(n0 + bcol) * K + bkq;
    unsigned ldsOff[2];
    ldsOff[0] = (unsigned)(uintptr_t)&ldsB[0][bcol * 32 + bkq];
    ldsOff[1] = (unsigned)(uintptr_t)&ldsB[1][bcol * 32 + bkq];

    f32x8 acc[4];
#pragma unroll
    for (int j = 0; j < 4; ++j)
#pragma unroll
        for (int i = 0; i < 8; ++i) acc[j][i] = 0.f;

    const __bf16* Abase = A + (size_t)(mrow0 + l16) * K + hi8;

    int nsteps = K >> 5;
    async_load_b128(ldsOff[0], gB);                     // prologue: tile 0

    for (int s = 0; s < nsteps; ++s) {
        int p = s & 1;
        if (s + 1 < nsteps) {
            async_load_b128(ldsOff[p ^ 1], gB + (s + 1) * 32);
            wait_async_le(1);                           // tile s complete (this wave)
        } else {
            wait_async_le(0);
        }
        __syncthreads();                                // tile s complete (all waves)

        if (s + 8 < nsteps) __builtin_prefetch(Abase + (s + 8) * 32, 0, 1);
        bf16x16 af = load_frag(Abase + s * 32);
#pragma unroll
        for (int j = 0; j < 4; ++j) {
            bf16x16 bfr = load_frag(&ldsB[p][(j * 16 + l16) * 32 + hi8]);
            acc[j] = wmma_bf16(af, bfr, acc[j]);
        }
        __syncthreads();                                // reads done before rewrite
    }

#pragma unroll
    for (int j = 0; j < 4; ++j) {
#pragma unroll
        for (int i = 0; i < 8; ++i) {
            int m = mrow0 + i + hi8;               // C layout: rows split 0-7 / 8-15
            int col = n0 + j * 16 + l16;
            float v = acc[j][i];
            if (bias) v += bias[col];
            if (mode == 2) {
                v = 0.5f * v * (1.0f + erff(v * 0.70710678118654752f));
                outB[(size_t)m * Nc + col] = tobf(v);
            } else if (mode == 3) {
                int b = m / N_;
                float gate = ada[(size_t)b * ADA_LD + gate_off + col];
                outF[(size_t)m * Nc + col] =
                    resid[(size_t)m * Nc + col] + gate * v;
            } else {
                outF[(size_t)m * Nc + col] = v;
            }
        }
    }
}

// ------------------------------------------- RoPE + layout to [B,H,N,64] bf16
__global__ void k_rope(const float* __restrict__ pre, __bf16* __restrict__ dst,
                       int H) {
    int idx = blockIdx.x * 256 + threadIdx.x;   // B*N*H*64
    int d = idx & 63;
    int h = (idx >> 6) % H;
    int n = ((idx >> 6) / H) % N_;
    int b = idx / (64 * H * N_);
    const float* prow = pre + (((size_t)(b * N_ + n) * H + h) << 6);
    float v  = prow[d];
    float vr = (d < 32) ? -prow[d + 32] : prow[d - 32];
    float inv = __expf(-((float)(d & 31) / 32.0f) * 9.2103403719761836f); // ln 1e4
    float ang = (float)n * inv;
    float s, c;
    __sincosf(ang, &s, &c);
    dst[((((size_t)b * H + h) * N_ + n) << 6) + d] = tobf(v * c + vr * s);
}

// ------------------------------------- V to [B,HKV,64,N] (d-major) bf16
__global__ void k_vperm(const float* __restrict__ pre, __bf16* __restrict__ vt) {
    int idx = blockIdx.x * 256 + threadIdx.x;   // B*N*HKV*64
    int d = idx & 63;
    int h = (idx >> 6) % HKV_;
    int n = ((idx >> 6) / HKV_) % N_;
    int b = idx / (64 * HKV_ * N_);
    vt[((((size_t)b * HKV_ + h) << 6) + d) * N_ + n] = tobf(pre[idx]);
}

// --------------------------------------------- flash attention, WMMA bf16
// one wave per 16-query tile; 32 keys / iteration; online softmax
__global__ void k_attn(const __bf16* __restrict__ Qb,
                       const __bf16* __restrict__ Kb,
                       const __bf16* __restrict__ Vt,
                       __bf16* __restrict__ ctx) {
    __shared__ __align__(16) __bf16 lds[4][16 * 32];
    int lane = threadIdx.x & 31;
    int wave = threadIdx.x >> 5;
    int gt = blockIdx.x * 4 + wave;     // [0, B*HQ*128)
    int qt = gt & 127;
    int h  = (gt >> 7) & 15;
    int b  = gt >> 11;
    int hk = h >> 2;                    // GQA group
    int l16 = lane & 15, hi8 = (lane >> 4) << 3;

    const __bf16* Qrow =
        Qb + (((size_t)(b * HQ_ + h) * N_ + qt * 16 + l16) << 6) + hi8;
    bf16x16 qf0 = load_frag(Qrow);
    bf16x16 qf1 = load_frag(Qrow + 32);

    const __bf16* Kh = Kb + (((size_t)(b * HKV_ + hk) * N_) << 6);
    const __bf16* Vh = Vt + (((size_t)(b * HKV_ + hk)) << 6) * N_;

    f32x8 o[4];
    float mrun[8], srun[8];
#pragma unroll
    for (int i = 0; i < 8; ++i) { mrun[i] = -1e30f; srun[i] = 0.f; }
#pragma unroll
    for (int j = 0; j < 4; ++j)
#pragma unroll
        for (int i = 0; i < 8; ++i) o[j][i] = 0.f;

    __bf16* P = lds[wave];

    for (int kb = 0; kb < N_; kb += 32) {
        f32x8 s0, s1;
#pragma unroll
        for (int i = 0; i < 8; ++i) { s0[i] = 0.f; s1[i] = 0.f; }

        const __bf16* K0 = Kh + ((size_t)(kb + l16) << 6) + hi8;
        bf16x16 kf = load_frag(K0);
        s0 = wmma_bf16(qf0, kf, s0);
        kf = load_frag(K0 + 32);
        s0 = wmma_bf16(qf1, kf, s0);
        const __bf16* K1 = K0 + (16 << 6);
        kf = load_frag(K1);
        s1 = wmma_bf16(qf0, kf, s1);
        kf = load_frag(K1 + 32);
        s1 = wmma_bf16(qf1, kf, s1);

        // online softmax over the 32 keys; rows live in 16-lane halves
#pragma unroll
        for (int i = 0; i < 8; ++i) {
            float a = s0[i] * 0.125f;     // 1/sqrt(64)
            float c = s1[i] * 0.125f;
            float rm = fmaxf(a, c);
            rm = fmaxf(rm, __shfl_xor(rm, 1, 16));
            rm = fmaxf(rm, __shfl_xor(rm, 2, 16));
            rm = fmaxf(rm, __shfl_xor(rm, 4, 16));
            rm = fmaxf(rm, __shfl_xor(rm, 8, 16));
            float mnew = fmaxf(mrun[i], rm);
            float scale = __expf(mrun[i] - mnew);
            float p0 = __expf(a - mnew), p1 = __expf(c - mnew);
            float ls = p0 + p1;
            ls += __shfl_xor(ls, 1, 16);
            ls += __shfl_xor(ls, 2, 16);
            ls += __shfl_xor(ls, 4, 16);
            ls += __shfl_xor(ls, 8, 16);
            srun[i] = srun[i] * scale + ls;
            mrun[i] = mnew;
#pragma unroll
            for (int j = 0; j < 4; ++j) o[j][i] *= scale;
            int m = i + hi8;
            P[m * 32 + l16]      = tobf(p0);
            P[m * 32 + 16 + l16] = tobf(p1);
        }

        asm volatile("s_wait_dscnt 0" ::: "memory");   // DS RAW before re-read
        bf16x16 pf = load_frag(P + l16 * 32 + hi8);    // C-layout -> A-layout

#pragma unroll
        for (int j = 0; j < 4; ++j) {
            const __bf16* Vp = Vh + (size_t)(j * 16 + l16) * N_ + kb + hi8;
            bf16x16 vf = load_frag(Vp);
            o[j] = wmma_bf16(pf, vf, o[j]);
        }
    }

#pragma unroll
    for (int i = 0; i < 8; ++i) {
        float inv = 1.0f / srun[i];
        int n = qt * 16 + i + hi8;
#pragma unroll
        for (int j = 0; j < 4; ++j) {
            ctx[((size_t)(b * N_ + n)) * D_ + h * 64 + j * 16 + l16] =
                tobf(o[j][i] * inv);
        }
    }
}

// =========================================================================
extern "C" void kernel_launch(void* const* d_in, const int* in_sizes, int n_in,
                              void* d_out, int out_size, void* d_ws, size_t ws_size,
                              hipStream_t stream) {
    (void)in_sizes; (void)n_in; (void)out_size; (void)ws_size;
    const float* x     = (const float*)d_in[0];
    const float* t_emb = (const float*)d_in[1];
    const float* Wq    = (const float*)d_in[2];
    const float* Wk    = (const float*)d_in[3];
    const float* Wv    = (const float*)d_in[4];
    const float* Wo    = (const float*)d_in[5];
    const float* W1    = (const float*)d_in[6];
    const float* b1    = (const float*)d_in[7];
    const float* W2    = (const float*)d_in[8];
    const float* b2    = (const float*)d_in[9];
    const float* Wada  = (const float*)d_in[10];
    const float* bada  = (const float*)d_in[11];
    const float* g1v   = (const float*)d_in[12];
    const float* g2v   = (const float*)d_in[13];

    char* wp = (char*)d_ws;
    auto alloc = [&](size_t bytes) -> void* {
        void* p = (void*)wp;
        wp += (bytes + 255) & ~(size_t)255;
        return p;
    };

    float*  ada  = (float*)alloc((size_t)B_ * ADA_LD * 4);
    __bf16* xn   = (__bf16*)alloc((size_t)MTOK * D_ * 2);
    __bf16* WqT  = (__bf16*)alloc((size_t)D_ * D_ * 2);
    __bf16* WkT  = (__bf16*)alloc((size_t)D_ * 256 * 2);
    __bf16* WvT  = (__bf16*)alloc((size_t)D_ * 256 * 2);
    __bf16* WoT  = (__bf16*)alloc((size_t)D_ * D_ * 2);
    __bf16* W1T  = (__bf16*)alloc((size_t)D_ * 4 * D_ * 2);
    __bf16* W2T  = (__bf16*)alloc((size_t)4 * D_ * D_ * 2);
    float*  Qpre = (float*)alloc((size_t)MTOK * D_ * 4);
    float*  Kpre = (float*)alloc((size_t)MTOK * 256 * 4);
    float*  Vpre = (float*)alloc((size_t)MTOK * 256 * 4);
    __bf16* Qbf  = (__bf16*)alloc((size_t)B_ * HQ_ * N_ * 64 * 2);
    __bf16* Kbf  = (__bf16*)alloc((size_t)B_ * HKV_ * N_ * 64 * 2);
    __bf16* Vtb  = (__bf16*)alloc((size_t)B_ * HKV_ * N_ * 64 * 2);
    __bf16* ctx  = (__bf16*)alloc((size_t)MTOK * D_ * 2);
    float*  x1   = (float*)alloc((size_t)MTOK * D_ * 4);
    __bf16* hbuf = (__bf16*)alloc((size_t)MTOK * 4 * D_ * 2);

    dim3 blk(256);

    // weight conversion (fp32 -> transposed bf16)
    k_cvt_t<<<(D_ * D_ + 255) / 256, blk, 0, stream>>>(Wq, WqT, D_, D_);
    k_cvt_t<<<(D_ * 256 + 255) / 256, blk, 0, stream>>>(Wk, WkT, D_, 256);
    k_cvt_t<<<(D_ * 256 + 255) / 256, blk, 0, stream>>>(Wv, WvT, D_, 256);
    k_cvt_t<<<(D_ * D_ + 255) / 256, blk, 0, stream>>>(Wo, WoT, D_, D_);
    k_cvt_t<<<(D_ * 4 * D_ + 255) / 256, blk, 0, stream>>>(W1, W1T, D_, 4 * D_);
    k_cvt_t<<<(4 * D_ * D_ + 255) / 256, blk, 0, stream>>>(W2, W2T, 4 * D_, D_);

    // adaLN parameters
    k_ada<<<(B_ * ADA_LD) / 256, blk, 0, stream>>>(t_emb, Wada, bada, ada);

    // norm1 + modulate
    k_norm_mod<<<MTOK, blk, 0, stream>>>(x, g1v, ada, /*scale*/ D_, /*shift*/ 0, xn);

    // QKV projections (f32 pre-RoPE)
    dim3 gQ(D_ / 64, MTOK / 128);
    k_gemm<<<gQ, blk, 0, stream>>>(xn, WqT, D_, D_, Qpre, nullptr, nullptr,
                                   nullptr, nullptr, 0, 0);
    dim3 gKV(256 / 64, MTOK / 128);
    k_gemm<<<gKV, blk, 0, stream>>>(xn, WkT, D_, 256, Kpre, nullptr, nullptr,
                                    nullptr, nullptr, 0, 0);
    k_gemm<<<gKV, blk, 0, stream>>>(xn, WvT, D_, 256, Vpre, nullptr, nullptr,
                                    nullptr, nullptr, 0, 0);

    // RoPE + layout transforms
    k_rope<<<(B_ * N_ * HQ_ * 64) / 256, blk, 0, stream>>>(Qpre, Qbf, HQ_);
    k_rope<<<(B_ * N_ * HKV_ * 64) / 256, blk, 0, stream>>>(Kpre, Kbf, HKV_);
    k_vperm<<<(B_ * N_ * HKV_ * 64) / 256, blk, 0, stream>>>(Vpre, Vtb);

    // attention
    k_attn<<<(B_ * HQ_ * (N_ / 16)) / 4, dim3(128), 0, stream>>>(Qbf, Kbf, Vtb, ctx);

    // x1 = x + gate_msa * (ctx @ Wo)
    k_gemm<<<gQ, blk, 0, stream>>>(ctx, WoT, D_, D_, x1, nullptr, nullptr,
                                   x, ada, 2 * D_, 3);

    // norm2 + modulate
    k_norm_mod<<<MTOK, blk, 0, stream>>>(x1, g2v, ada, /*scale*/ 4 * D_,
                                         /*shift*/ 3 * D_, xn);

    // h = gelu(xn @ W1 + b1)
    dim3 gH(4 * D_ / 64, MTOK / 128);
    k_gemm<<<gH, blk, 0, stream>>>(xn, W1T, D_, 4 * D_, nullptr, hbuf, b1,
                                   nullptr, nullptr, 0, 2);

    // out = x1 + gate_mlp * (h @ W2 + b2)
    k_gemm<<<gQ, blk, 0, stream>>>(hbuf, W2T, 4 * D_, D_, (float*)d_out,
                                   nullptr, b2, x1, ada, 5 * D_, 3);
}
